// CrystalLinear_45389214384637
// MI455X (gfx1250) — matrix operand order repaired and checked by reading, not compiled, not run on hardware
//
#include <hip/hip_runtime.h>
#include <stdint.h>

typedef __attribute__((ext_vector_type(16))) _Float16 v16h;
typedef __attribute__((ext_vector_type(8)))  _Float16 v8h;
typedef __attribute__((ext_vector_type(8)))  float    v8f;

constexpr int MDIM = 1024;
constexpr int KDIM = 8192;
constexpr int NDIM = 8192;
constexpr int KW   = KDIM / 16;   // packed dwords per weight row
constexpr int BM   = 128;
constexpr int BN   = 128;
constexpr int KS   = 32;          // K per WMMA step
constexpr int KT   = KDIM / KS;   // 256 K-steps
constexpr int XPAD = 40;          // halves per LDS row (32 data + 8 pad), 80B stride
constexpr int NBUF = 4;           // power-of-2 buffer ring: index is (kt & 3)

union V16HU { v16h v; uint32_t u[8]; };

__global__ __launch_bounds__(256)
void ternary_wmma_gemm(const _Float16* __restrict__ x,
                       const uint32_t* __restrict__ pw,
                       const _Float16* __restrict__ bias,
                       _Float16* __restrict__ y) {
  __shared__ _Float16 xs[NBUF][BM][XPAD];   // 4-deep X-tile ring (40 KB)

  const int tid  = threadIdx.x;
  const int lane = tid & 31;
  const int wave = tid >> 5;
  const int wm   = wave & 1;      // 64-row slab within block
  const int wn   = wave >> 1;     // 32-col slab within block
  const int m0   = blockIdx.y * BM;
  const int n0   = blockIdx.x * BN;
  const int l15  = lane & 15;
  const int lhi  = lane >> 4;     // 0 or 1

  // --- async X-tile loader: global -> LDS, tracked by ASYNCcnt (2 instr/thread) ---
  auto load_x = [&](int b, int k0) {
#pragma unroll
    for (int it = 0; it < 2; ++it) {
      int i   = tid + it * 256;
      int row = i >> 2;           // 0..127
      int seg = (i & 3) * 8;      // half-offset 0/8/16/24 (16B chunks)
      const _Float16* gp = x + (size_t)(m0 + row) * KDIM + (k0 + seg);
      uint32_t loff = (uint32_t)(uintptr_t)&xs[b][row][seg];
      asm volatile("global_load_async_to_lds_b128 %0, %1, off"
                   :: "v"(loff), "v"(gp) : "memory");
    }
  };

  v8f acc[4][2];
  {
    v8f z = {};
#pragma unroll
    for (int mt = 0; mt < 4; ++mt)
#pragma unroll
      for (int nt = 0; nt < 2; ++nt) acc[mt][nt] = z;
  }

  // fp16 high-byte LUT for ternary codes 0,1,2,3 -> 0.0,1.0,2.0,3.0 (code 2 unused)
  const uint32_t lut = 0x42403C00u;
  const int nbase = n0 + wn * 32;
  const uint32_t* pw0 = pw + (size_t)(nbase + l15) * KW + lhi;  // nt=0 column
  const uint32_t* pw1 = pw0 + (size_t)16 * KW;                  // nt=1 column

  // --- pipeline prologue: 2 LDS stages in flight, stage-0 B words in regs ---
  load_x(0, 0);
  load_x(1, KS);
  uint32_t wcur0 = pw0[0];
  uint32_t wcur1 = pw1[0];
  asm volatile("s_wait_asynccnt 0x2" ::: "memory");  // stage 0 landed
  __syncthreads();

  for (int kt = 0; kt < KT; ++kt) {
    const int buf = kt & (NBUF - 1);
    const bool more2 = (kt + 2 < KT);
    if (more2) load_x((kt + 2) & (NBUF - 1), (kt + 2) * KS);

    // prefetch next iteration's packed B words (index clamped; value unused on tail)
    const int ktn = (kt + 1 < KT) ? (kt + 1) : kt;
    uint32_t wnx0 = pw0[ktn * 2];
    uint32_t wnx1 = pw1[ktn * 2];

    // --- decode B fragments: one packed dword = lane's 16 consecutive K codes ---
    uint32_t w[2] = {wcur0, wcur1};
    v16h bf[2];
#pragma unroll
    for (int nt = 0; nt < 2; ++nt) {
      V16HU t;
#pragma unroll
      for (int j = 0; j < 8; ++j) {
        uint32_t c0 = (w[nt] >> (4 * j)) & 3u;
        uint32_t c1 = (w[nt] >> (4 * j + 2)) & 3u;
        // result bytes: [0]=lut[0]=0, [1]=lut[c0], [2]=0, [3]=lut[c1] -> two f16
        t.u[j] = __builtin_amdgcn_perm(0u, lut, (c0 << 8) | (c1 << 24));
      }
      bf[nt] = t.v;
    }

    // --- A fragments from LDS ---
    // 16-bit A 16x32 layout (matches B decode): lanes0-15 hold K0-15;
    // lanes16-31 hold K16-31.
    v16h af[4];
#pragma unroll
    for (int mt = 0; mt < 4; ++mt) {
      const _Float16* rp = &xs[buf][wm * 64 + mt * 16 + l15][lhi * 16];
      v8h lo = *(const v8h*)(rp);        // K koff..koff+7
      v8h hi = *(const v8h*)(rp + 8);    // K koff+8..koff+15
      af[mt] = __builtin_shufflevector(lo, hi,
                 0, 1, 2, 3, 4, 5, 6, 7, 8, 9, 10, 11, 12, 13, 14, 15);
    }

    // --- 8 WMMAs per wave per K-step ---
#pragma unroll
    for (int mt = 0; mt < 4; ++mt)
#pragma unroll
      for (int nt = 0; nt < 2; ++nt)
        acc[mt][nt] = __builtin_amdgcn_wmma_f32_16x16x32_f16(
            false, af[mt], false, bf[nt], (short)0, acc[mt][nt], false, false);

    wcur0 = wnx0;
    wcur1 = wnx1;

    // async loads retire in order: <=2 outstanding guarantees stage kt+1 landed
    if (more2)
      asm volatile("s_wait_asynccnt 0x2" ::: "memory");
    else
      asm volatile("s_wait_asynccnt 0x0" ::: "memory");
    __syncthreads();
  }

  // --- epilogue: C layout VGPR r -> rows r (lanes 0-15) / r+8 (lanes 16-31) ---
  _Float16 bv[2];
#pragma unroll
  for (int nt = 0; nt < 2; ++nt) bv[nt] = bias[nbase + nt * 16 + l15];

#pragma unroll
  for (int mt = 0; mt < 4; ++mt) {
    const int mrow = m0 + wm * 64 + mt * 16 + lhi * 8;
#pragma unroll
    for (int nt = 0; nt < 2; ++nt) {
      const int nc = nbase + nt * 16 + l15;
#pragma unroll
      for (int r = 0; r < 8; ++r) {
        y[(size_t)(mrow + r) * NDIM + nc] = (_Float16)acc[mt][nt][r] + bv[nt];
      }
    }
  }
}

extern "C" void kernel_launch(void* const* d_in, const int* in_sizes, int n_in,
                              void* d_out, int out_size, void* d_ws, size_t ws_size,
                              hipStream_t stream) {
  const _Float16*  x    = (const _Float16*)d_in[0];
  const uint32_t*  pwv  = (const uint32_t*)d_in[1];
  const _Float16*  bias = (const _Float16*)d_in[2];
  _Float16*        y    = (_Float16*)d_out;

  dim3 grid(NDIM / BN, MDIM / BM);   // (64, 8)
  dim3 block(256);
  ternary_wmma_gemm<<<grid, block, 0, stream>>>(x, pwv, bias, y);
}
